// MoeExperts4bit_69346541961366
// MI455X (gfx1250) — compile-verified, matrix operand order checked by reading, not run on hardware
//
#include <hip/hip_runtime.h>

// ---------------- problem constants (match reference) ----------------
#define T_TOK 1024
#define K_TOP 2
#define E_EXP 8
#define H_DIM 2048
#define I_DIM 768
#define TK (T_TOK * K_TOP)      // 2048 routed (token, expert) pairs

// ---------------- CDNA5 WMMA types ----------------
typedef __attribute__((ext_vector_type(16))) __bf16 v16bf;
typedef __attribute__((ext_vector_type(8)))  float  v8f;
typedef __attribute__((ext_vector_type(4)))  int    v4i_t;

// pointer types matching the async-to-LDS builtin's parameters
typedef __attribute__((address_space(1))) v4i_t* g128_t;   // global int4*
typedef __attribute__((address_space(3))) v4i_t* l128_t;   // LDS    int4*

union FragU { v16bf v; uint4 u[2]; };

// round-to-nearest-even fp32 -> bf16
__device__ __forceinline__ unsigned short f2bf(float f) {
    unsigned u = __builtin_bit_cast(unsigned, f);
    u += 0x7FFFu + ((u >> 16) & 1u);
    return (unsigned short)(u >> 16);
}
__device__ __forceinline__ unsigned pack2(float a, float b) {
    return (unsigned)f2bf(a) | ((unsigned)f2bf(b) << 16);
}

// ---------------- async global -> LDS staging (CDNA5 path) ----------------
#if __has_builtin(__builtin_amdgcn_global_load_async_to_lds_b128)
#define HAS_ASYNC_LDS 1
#else
#define HAS_ASYNC_LDS 0
#endif

// Copy 16 bytes (8 bf16) from global to LDS. Async DMA when available
// (ASYNCcnt-tracked, no VGPR round trip); plain copy otherwise.
__device__ __forceinline__ void stage16(const unsigned short* g, unsigned short* l) {
#if HAS_ASYNC_LDS
    __builtin_amdgcn_global_load_async_to_lds_b128(
        (g128_t)(unsigned short*)g,   // global src (AS1 int4*)
        (l128_t)l,                    // LDS dst    (AS3 int4*)
        0, 0);
#else
    *(uint4*)l = *(const uint4*)g;
#endif
}

__device__ __forceinline__ void wait_async() {
#if HAS_ASYNC_LDS
#if __has_builtin(__builtin_amdgcn_s_wait_asynccnt)
    __builtin_amdgcn_s_wait_asynccnt(0);
#else
    asm volatile("s_wait_asynccnt 0x0" ::: "memory");
#endif
#endif
}

// Load one 16x32 bf16 WMMA operand fragment from an LDS tile (row stride = 32
// bf16 = 64B). ISA 7.12.2 (16-bit A 16x32): lane L holds row (L&15); lanes
// 0-15 carry K=0..7 & 16..23, lanes 16-31 carry K=8..15 & 24..31, two
// memory-contiguous 8-element runs -> two ds_load_b128.
__device__ __forceinline__ v16bf load_frag(const unsigned short* tile, int lane) {
    const int row  = lane & 15;
    const int half = lane >> 4;              // 0 or 1
    const unsigned short* rp = tile + row * 32 + half * 8;
    FragU f;
    f.u[0] = *(const uint4*)(rp);            // K = half*8 .. +7
    f.u[1] = *(const uint4*)(rp + 16);       // K = 16 + half*8 .. +7
    return f.v;
}

// ---------------- kernel 0: streaming fp32 -> bf16 convert ----------------
// One-time pass; the bf16 mirror (~80 MB) then lives in the 192 MB L2.
__global__ void cvt_f32_bf16(const float* __restrict__ src,
                             unsigned short* __restrict__ dst, int n8) {
    int i = blockIdx.x * blockDim.x + threadIdx.x;   // i indexes groups of 8
    if (i >= n8) return;
    const float4* s = (const float4*)src + (size_t)2 * i;
    float4 a = s[0], b = s[1];
    uint4 o;
    o.x = pack2(a.x, a.y); o.y = pack2(a.z, a.w);
    o.z = pack2(b.x, b.y); o.w = pack2(b.z, b.w);
    ((uint4*)dst)[i] = o;
}

// ---------------- kernel 1: routing scatter ----------------
__global__ void moe_route(const int* __restrict__ topk,
                          int* __restrict__ counts,
                          int* __restrict__ plist) {
    int p = blockIdx.x * blockDim.x + threadIdx.x;
    if (p >= TK) return;
    int e = topk[p];
    int pos = atomicAdd(&counts[e], 1);
    plist[e * TK + pos] = p;                 // pair id; token = p >> 1
}

// ---------------- kernel 2: grouped GEMM1 + silu*up*w -> h (bf16) ----------
// grid: (I/64, TK/16, E), block: 128 threads = 4 waves.
// Each wave owns one 16x16 gate tile and one 16x16 up tile.
__global__ __launch_bounds__(128)
void moe_gemm1(const unsigned short* __restrict__ xbf,   // [T,H] bf16
               const float* __restrict__ wtop,           // [T,K] flat
               const unsigned short* __restrict__ gubf,  // [E,2I,H] bf16
               const int*   __restrict__ counts,         // [E]
               const int*   __restrict__ plist,          // [E,TK]
               unsigned short* __restrict__ hbuf)        // [TK,I] bf16
{
    __shared__ __align__(16) unsigned short sA[16 * 32];
    __shared__ __align__(16) unsigned short sBg[64 * 32];
    __shared__ __align__(16) unsigned short sBu[64 * 32];
    __shared__ int   sPair[16];
    __shared__ int   sTok[16];
    __shared__ float sW[16];

    const int e  = blockIdx.z;
    const int n  = counts[e];
    const int rt = blockIdx.y;               // 16-row tile of this expert's list
    if (rt * 16 >= n) return;
    const int ct   = blockIdx.x;             // 64-col tile of I
    const int tid  = threadIdx.x;
    const int lane = tid & 31;
    const int wave = tid >> 5;

    if (tid < 16) {
        int idx  = rt * 16 + tid;
        int cidx = (idx < n) ? idx : (n - 1);
        int p    = plist[e * TK + cidx];
        sPair[tid] = (idx < n) ? p : -1;
        sTok[tid]  = p >> 1;                 // K_TOP == 2
        sW[tid]    = wtop[p];
    }
    __syncthreads();

    const unsigned short* wg = gubf + ((size_t)e * (2 * I_DIM) + (size_t)ct * 64) * H_DIM;
    const unsigned short* wu = gubf + ((size_t)e * (2 * I_DIM) + I_DIM + (size_t)ct * 64) * H_DIM;

    v8f accG = {};
    v8f accU = {};

    for (int kk = 0; kk < H_DIM; kk += 32) {
        // ---- stage A: 16 gathered token rows x 32 bf16 (threads 0..63)
        if (tid < 64) {
            int r = tid >> 2;
            int c = (tid & 3) * 8;
            stage16(xbf + (size_t)sTok[r] * H_DIM + kk + c, &sA[r * 32 + c]);
        }
        // ---- stage B: 64 gate rows + 64 up rows x 32 bf16 (thread = row)
        {
            const unsigned short* src = (tid < 64)
                ? (wg + (size_t)tid * H_DIM + kk)
                : (wu + (size_t)(tid - 64) * H_DIM + kk);
            unsigned short* dst = (tid < 64) ? &sBg[tid * 32] : &sBu[(tid - 64) * 32];
#pragma unroll
            for (int j = 0; j < 32; j += 8) stage16(src + j, dst + j);
        }
        wait_async();
        __syncthreads();

        v16bf fa = load_frag(sA, lane);
        v16bf fg = load_frag(&sBg[wave * 16 * 32], lane);
        v16bf fu = load_frag(&sBu[wave * 16 * 32], lane);
        accG = __builtin_amdgcn_wmma_f32_16x16x32_bf16(false, fa, false, fg,
                                                       (short)0, accG, false, false);
        accU = __builtin_amdgcn_wmma_f32_16x16x32_bf16(false, fa, false, fu,
                                                       (short)0, accU, false, false);
        __syncthreads();
    }

    // ---- epilogue: h = silu(gate)*up * w, store bf16 to hbuf[pair, col]
    const int half = lane >> 4;
    const int nn   = lane & 15;
    const int col  = ct * 64 + wave * 16 + nn;
#pragma unroll
    for (int r = 0; r < 8; ++r) {
        int m = r + half * 8;                // C/D layout: VGPR r, half-wave
        int p = sPair[m];
        if (p >= 0) {
            float g = accG[r], u = accU[r];
            float sig = __builtin_amdgcn_rcpf(1.0f + __expf(-g));  // v_rcp_f32
            float hv = g * sig * u * sW[m];
            hbuf[(size_t)p * I_DIM + col] = f2bf(hv);
        }
    }
}

// ---------------- kernel 3: grouped GEMM2 (h @ Wd^T) -> per-pair rows ------
// grid: (H/64, TK/16, E), block: 128 threads = 4 waves.
__global__ __launch_bounds__(128)
void moe_gemm2(const unsigned short* __restrict__ hbuf, // [TK,I] bf16
               const unsigned short* __restrict__ dbf,  // [E,H,I] bf16
               const int*   __restrict__ counts,
               const int*   __restrict__ plist,
               float* __restrict__ pout)                // [TK,H] f32
{
    __shared__ __align__(16) unsigned short sA[16 * 32];
    __shared__ __align__(16) unsigned short sB[64 * 32];
    __shared__ int sPair[16];
    __shared__ int sRow[16];

    const int e  = blockIdx.z;
    const int n  = counts[e];
    const int rt = blockIdx.y;
    if (rt * 16 >= n) return;
    const int ct   = blockIdx.x;             // 64-col tile of H
    const int tid  = threadIdx.x;
    const int lane = tid & 31;
    const int wave = tid >> 5;

    if (tid < 16) {
        int idx  = rt * 16 + tid;
        int cidx = (idx < n) ? idx : (n - 1);
        int p    = plist[e * TK + cidx];
        sPair[tid] = (idx < n) ? p : -1;
        sRow[tid]  = p;                      // clamped, always a valid hbuf row
    }
    __syncthreads();

    const unsigned short* wd = dbf + ((size_t)e * H_DIM + (size_t)ct * 64) * I_DIM;

    v8f acc = {};

    for (int kk = 0; kk < I_DIM; kk += 32) {
        // ---- stage A: 16 gathered bf16 h rows (threads 0..63)
        if (tid < 64) {
            int r = tid >> 2;
            int c = (tid & 3) * 8;
            stage16(hbuf + (size_t)sRow[r] * I_DIM + kk + c, &sA[r * 32 + c]);
        }
        // ---- stage B: 64 rows of Wd (bf16), half row per thread
        {
            int r   = tid >> 1;
            int seg = (tid & 1) * 16;
            const unsigned short* src = wd + (size_t)r * I_DIM + kk + seg;
            stage16(src,     &sB[r * 32 + seg]);
            stage16(src + 8, &sB[r * 32 + seg + 8]);
        }
        wait_async();
        __syncthreads();

        v16bf fa = load_frag(sA, lane);
        v16bf fb = load_frag(&sB[wave * 16 * 32], lane);
        acc = __builtin_amdgcn_wmma_f32_16x16x32_bf16(false, fa, false, fb,
                                                      (short)0, acc, false, false);
        __syncthreads();
    }

    const int half = lane >> 4;
    const int nn   = lane & 15;
    const int col  = ct * 64 + wave * 16 + nn;
#pragma unroll
    for (int r = 0; r < 8; ++r) {
        int m = r + half * 8;
        int p = sPair[m];
        if (p >= 0) pout[(size_t)p * H_DIM + col] = acc[r];
    }
}

// ---------------- kernel 4: deterministic combine over K=2 pairs ----------
__global__ void moe_combine(const float* __restrict__ pout,  // [TK,H]
                            float* __restrict__ out)         // [T,H]
{
    int i = blockIdx.x * blockDim.x + threadIdx.x;           // over T*H
    if (i >= T_TOK * H_DIM) return;
    int t = i / H_DIM;
    int h = i - t * H_DIM;
    size_t base = (size_t)(2 * t) * H_DIM + h;
    out[i] = pout[base] + pout[base + H_DIM];
}

// ---------------- host launcher ----------------
extern "C" void kernel_launch(void* const* d_in, const int* in_sizes, int n_in,
                              void* d_out, int out_size, void* d_ws, size_t ws_size,
                              hipStream_t stream) {
    const float* x    = (const float*)d_in[0];   // hidden_states [T,H]
    const int*   topk = (const int*)  d_in[1];   // top_k_index  [T,K]
    const float* wtop = (const float*)d_in[2];   // top_k_weights[T,K]
    const float* gup  = (const float*)d_in[3];   // gate_up_proj [E,2I,H]
    const float* dwn  = (const float*)d_in[4];   // down_proj    [E,H,I]
    float* out = (float*)d_out;

    // ---- workspace layout (offsets in bytes, all 256-aligned) ----
    char* ws = (char*)d_ws;
    size_t off = 0;
    int* counts = (int*)(ws + off);                    off += 256;
    int* plist  = (int*)(ws + off);                    off += (size_t)E_EXP * TK * 4;       // 64 KB
    unsigned short* xbf  = (unsigned short*)(ws + off); off += (size_t)T_TOK * H_DIM * 2;   // 4 MB
    unsigned short* gubf = (unsigned short*)(ws + off); off += (size_t)E_EXP * 2 * I_DIM * H_DIM * 2; // 48 MB
    unsigned short* dbf  = (unsigned short*)(ws + off); off += (size_t)E_EXP * H_DIM * I_DIM * 2;     // 24 MB
    unsigned short* hbuf = (unsigned short*)(ws + off); off += (size_t)TK * I_DIM * 2;      // 3 MB
    float* pout = (float*)(ws + off);                  off += (size_t)TK * H_DIM * 4;       // 16 MB

    (void)hipMemsetAsync(counts, 0, E_EXP * sizeof(int), stream);

    // one-time fp32 -> bf16 mirrors (x, gate_up, down)
    {
        int n8x = T_TOK * H_DIM / 8;
        cvt_f32_bf16<<<(n8x + 255) / 256, 256, 0, stream>>>(x, xbf, n8x);
        int n8g = E_EXP * 2 * I_DIM * H_DIM / 8;
        cvt_f32_bf16<<<(n8g + 255) / 256, 256, 0, stream>>>(gup, gubf, n8g);
        int n8d = E_EXP * H_DIM * I_DIM / 8;
        cvt_f32_bf16<<<(n8d + 255) / 256, 256, 0, stream>>>(dwn, dbf, n8d);
    }

    moe_route<<<(TK + 255) / 256, 256, 0, stream>>>(topk, counts, plist);

    dim3 g1(I_DIM / 64, TK / 16, E_EXP);
    moe_gemm1<<<g1, 128, 0, stream>>>(xbf, wtop, gubf, counts, plist, hbuf);

    dim3 g2(H_DIM / 64, TK / 16, E_EXP);
    moe_gemm2<<<g2, 128, 0, stream>>>(hbuf, dbf, counts, plist, pout);

    moe_combine<<<(T_TOK * H_DIM + 255) / 256, 256, 0, stream>>>(pout, out);
}
